// intra_att_LR_61890478736013
// MI455X (gfx1250) — matrix operand order repaired and checked by reading, not compiled, not run on hardware
//
#include <hip/hip_runtime.h>
#include <hip/hip_bf16.h>

// Problem constants (match the JAX reference).
#define N_SRC 100000
#define MROWS 20000
#define KNEI  32
#define HDIM  128
#define NEG_SLOPE 0.01f

typedef __attribute__((ext_vector_type(16))) _Float16 v16h;
typedef __attribute__((ext_vector_type(4)))  _Float16 v4h;
typedef __attribute__((ext_vector_type(8)))  float    v8f;

// One wave32 handles one reference row m (its 32 neighbors).
// 8 waves per block; MROWS == 8 * 2500 exactly, so every wave is active and
// __syncthreads() is uniform.
__global__ __launch_bounds__(256) void intra_att_lr_kernel(
    const int*   __restrict__ nei,        // (M, K) int32
    const float* __restrict__ h,          // (N_SRC, 1)
    const float* __restrict__ h_refer,    // (M, 1)
    const float* __restrict__ map_l_w,    // (H, 1)
    const float* __restrict__ map_l_b,    // (H,)
    const float* __restrict__ map_r_w,    // (H, 1)
    const float* __restrict__ map_r_b,    // (H,)
    const float* __restrict__ att_inter,  // (1, 2H): [a_ref | a_nei]
    float* __restrict__ out,              // (M,)
    float* __restrict__ att)              // (M, K)
{
    __shared__ __align__(16) float    w_l_s[HDIM];        // shared weights (block-wide)
    __shared__ __align__(16) float    b_l_s[HDIM];
    __shared__ __align__(32) _Float16 anei_h[HDIM];       // a_nei in f16 (B column 1)
    __shared__ __align__(32) _Float16 hr_h[8][HDIM];      // per-wave hr in f16 (B column 0)
    __shared__               float    lr_s[8][KNEI];      // D column 0 exchange
    __shared__               float    ln_s[8][KNEI];      // D column 1 exchange

    const int tid  = threadIdx.x;
    const int wave = tid >> 5;
    const int lane = tid & 31;

    // ---- block init: stage the tiny weight vectors once ----
    if (tid < HDIM) {
        w_l_s[tid]  = map_l_w[tid];
        b_l_s[tid]  = map_l_b[tid];
        anei_h[tid] = (_Float16)att_inter[HDIM + tid];
    }
    __syncthreads();

    const int  m      = blockIdx.x * 8 + wave;
    const bool active = (m < MROWS);          // wave-uniform (and always true here)

    // Per-lane register constants for hr / g (4 contiguous j per lane).
    const int j0 = lane * 4;
    float wr[4], br[4], ar[4];
#pragma unroll
    for (int q = 0; q < 4; ++q) {
        wr[q] = map_r_w[j0 + q];
        br[q] = map_r_b[j0 + q];
        ar[q] = att_inter[j0 + q];            // a_ref
    }

    // ---- gather neighbor scalars, compute hr_m and g = <hr_m, a_ref> ----
    float x = 0.f, y = 0.f;
    if (active) {
        x = h[nei[m * KNEI + lane]];          // neighbor k = lane
        y = h_refer[m];
    }
    float g = 0.f;
    {
        v4h pack;
#pragma unroll
        for (int q = 0; q < 4; ++q) {
            float v = fmaxf(fmaf(y, wr[q], br[q]), 0.f);
            g += v * ar[q];
            pack[q] = (_Float16)v;
        }
        *(v4h*)&hr_h[wave][j0] = pack;        // stage hr in f16 for the B fragment
    }
#pragma unroll
    for (int off = 16; off > 0; off >>= 1) g += __shfl_xor(g, off, 32);
    __syncthreads();                          // hr_h / cross-lane visibility

    // ---- WMMA: D(32x16) = NE(32x128) @ [hr | a_nei | x...x](128x16) ----
    // A packing used here (contiguous k, same as B): lane holds row (lane&15); K base 0 (lanes<16) / 16
    // (lanes>=16); halves p -> K=base+p.
    // B packing used here: lane holds column (lane&15); halves p -> K = p + (lane>=16 ? 16 : 0).
    const int   col = lane & 15;
    const float xt0 = __shfl(x, col, 32);        // row scalar for tile 0 (k = col)
    const float xt1 = __shfl(x, col + 16, 32);   // row scalar for tile 1 (k = col+16)
    const int   khB = (lane >= 16) ? 16 : 0;
    const int   khA = (lane >= 16) ? 16 : 0;

    // Columns >= 2 of B are don't-care (columns of D are independent); point
    // them at hr_h so the load is always in-bounds.
    const _Float16* bsrc = (col == 1) ? anei_h : hr_h[wave];

    v8f acc0 = {};  // rows 0..15  (k = 0..15)
    v8f acc1 = {};  // rows 16..31 (k = 16..31)

#pragma unroll
    for (int c = 0; c < 4; ++c) {             // K chunks of 32 over H = 128
        v16h bfrag = *(const v16h*)&bsrc[c * 32 + khB];

        float wv[16], bv[16];
#pragma unroll
        for (int p = 0; p < 8; ++p) {
            const int j = c * 32 + khA + p;
            wv[p]     = w_l_s[j];       bv[p]     = b_l_s[j];
            wv[8 + p] = w_l_s[j + 8];  bv[8 + p] = b_l_s[j + 8];
        }
        v16h a0, a1;
#pragma unroll
        for (int p = 0; p < 16; ++p) {
            a0[p] = (_Float16)fmaxf(fmaf(xt0, wv[p], bv[p]), 0.f);
            a1[p] = (_Float16)fmaxf(fmaf(xt1, wv[p], bv[p]), 0.f);
        }
        acc0 = __builtin_amdgcn_wmma_f32_16x16x32_f16(false, a0, false, bfrag,
                                                      (short)0, acc0, false, false);
        acc1 = __builtin_amdgcn_wmma_f32_16x16x32_f16(false, a1, false, bfrag,
                                                      (short)0, acc1, false, false);
    }

    // ---- scatter D columns 0 (lr_inner) and 1 (logit_nei) to lanes by k ----
    // D layout: lane holds column (lane&15); VGPR v -> row v (+8 for lanes>=16).
    {
        const int r = (lane >= 16) ? 8 : 0;
        if (col == 0) {
#pragma unroll
            for (int v = 0; v < 8; ++v) {
                lr_s[wave][r + v]      = acc0[v];
                lr_s[wave][16 + r + v] = acc1[v];
            }
        } else if (col == 1) {
#pragma unroll
            for (int v = 0; v < 8; ++v) {
                ln_s[wave][r + v]      = acc0[v];
                ln_s[wave][16 + r + v] = acc1[v];
            }
        }
    }
    __syncthreads();

    // ---- leaky-relu -> softmax over K=32 (one lane per k) -> output ----
    const float lr = lr_s[wave][lane];
    float lg = g + ln_s[wave][lane];
    lg = (lg >= 0.f) ? lg : NEG_SLOPE * lg;

    float mx = lg;
#pragma unroll
    for (int off = 16; off > 0; off >>= 1) mx = fmaxf(mx, __shfl_xor(mx, off, 32));
    const float e = __expf(lg - mx);
    float s = e;
#pragma unroll
    for (int off = 16; off > 0; off >>= 1) s += __shfl_xor(s, off, 32);
    const float a = e / s;

    float o = a * lr;
#pragma unroll
    for (int off = 16; off > 0; off >>= 1) o += __shfl_xor(o, off, 32);

    if (active) {
        att[m * KNEI + lane] = a;
        if (lane == 0) out[m] = fmaxf(o, 0.f);
    }
}

extern "C" void kernel_launch(void* const* d_in, const int* in_sizes, int n_in,
                              void* d_out, int out_size, void* d_ws, size_t ws_size,
                              hipStream_t stream) {
    (void)in_sizes; (void)n_in; (void)out_size; (void)d_ws; (void)ws_size;
    const int*   nei       = (const int*)  d_in[0];
    const float* h         = (const float*)d_in[1];
    const float* h_refer   = (const float*)d_in[2];
    const float* map_l_w   = (const float*)d_in[3];
    const float* map_l_b   = (const float*)d_in[4];
    const float* map_r_w   = (const float*)d_in[5];
    const float* map_r_b   = (const float*)d_in[6];
    const float* att_inter = (const float*)d_in[7];

    float* out = (float*)d_out;           // (M,)   first
    float* att = out + MROWS;             // (M, K) second

    const int blocks = (MROWS + 7) / 8;   // 8 waves (rows) per 256-thread block
    intra_att_lr_kernel<<<blocks, 256, 0, stream>>>(
        nei, h, h_refer, map_l_w, map_l_b, map_r_w, map_r_b, att_inter, out, att);
}
